// SparseRKANBase_12962211299538
// MI455X (gfx1250) — compile-verified, matrix operand order checked by reading, not run on hardware
//
#include <hip/hip_runtime.h>
#include <hip/hip_bf16.h>
#include <stdint.h>

#define B_  64
#define T_  256
#define H_  2048
#define L_  2

typedef __attribute__((ext_vector_type(16))) __bf16 v16bf;
typedef __attribute__((ext_vector_type(8)))  __bf16 v8bf;
typedef __attribute__((ext_vector_type(8)))  float  v8f;
typedef __attribute__((ext_vector_type(4)))  float  f32x4;
typedef __attribute__((ext_vector_type(4)))  int    i32x4;

union V16U { v16bf v; v8bf h[2]; };

// ---------------- CDNA5 async global->LDS helpers -------------------------
__device__ __forceinline__ void async_load_b128(void* lds, const void* g) {
#if __has_builtin(__builtin_amdgcn_global_load_async_to_lds_b128)
  __builtin_amdgcn_global_load_async_to_lds_b128(
      (__attribute__((address_space(1))) i32x4*)g,
      (__attribute__((address_space(3))) i32x4*)lds, 0, 0);
#else
  asm volatile("global_load_async_to_lds_b128 %0, %1, off"
               :: "v"((unsigned)(unsigned long long)lds), "v"(g)
               : "memory");
#endif
}

__device__ __forceinline__ void wait_async0() {
#if __has_builtin(__builtin_amdgcn_s_wait_asynccnt)
  __builtin_amdgcn_s_wait_asynccnt(0);
#else
  asm volatile("s_wait_asynccnt 0" ::: "memory");
#endif
}

// ---------------------------------------------------------------------------
// Weight packing: masked f32 (L,H,H) [n][k] -> bf16 WMMA B-fragments.
//   chunk c = (((ls*64 + kt)*128 + nt)*2 + half)*32 + lane, 8 bf16 per chunk
//   n = nt*16 + lane%16 ; k = kt*32 + half*16 + (lane/16)*8 + j
// ---------------------------------------------------------------------------
__global__ __launch_bounds__(256) void pack_weights_k(
    const float* __restrict__ W_ih, const float* __restrict__ W_hh,
    const uint8_t* __restrict__ M_ih, const uint8_t* __restrict__ M_hh,
    __bf16* __restrict__ wp)
{
  unsigned cid = blockIdx.x * 256u + threadIdx.x;
  if (cid >= 4u * 64u * 128u * 2u * 32u) return;
  unsigned c = cid;
  unsigned lane = c & 31u;  c >>= 5;
  unsigned half = c & 1u;   c >>= 1;
  unsigned nt   = c & 127u; c >>= 7;
  unsigned kt   = c & 63u;  c >>= 6;
  unsigned ls   = c;
  const float*   W = ((ls & 1u) ? W_hh : W_ih) + (size_t)(ls >> 1) * H_ * H_;
  const uint8_t* M = ((ls & 1u) ? M_hh : M_ih) + (size_t)(ls >> 1) * H_ * H_;
  const int n  = nt * 16 + (lane & 15);
  const int k0 = kt * 32 + half * 16 + ((lane >> 4) * 8);
  v8bf o;
#pragma unroll
  for (int j = 0; j < 8; ++j) {
    const size_t idx = (size_t)n * H_ + k0 + j;
    float w = M[idx] ? W[idx] : 0.0f;
    o[j] = (__bf16)w;
  }
  *(v8bf*)(wp + (size_t)cid * 8) = o;
}

__global__ __launch_bounds__(256) void bias_k(
    const float* __restrict__ bi, const float* __restrict__ bh,
    float* __restrict__ bias)
{
  int i = blockIdx.x * 256 + threadIdx.x;
  if (i < L_ * H_) bias[i] = bi[i] + bh[i];
}

__global__ __launch_bounds__(256) void init_h_k(
    __bf16* h0, __bf16* h1, float* f0, float* f1)
{
  int i = blockIdx.x * 256 + threadIdx.x;
  if (i < B_ * H_) {
    h0[i] = (__bf16)0.0f; h1[i] = (__bf16)0.0f;
    f0[i] = 0.0f;         f1[i] = 0.0f;
  }
}

// ---------------------------------------------------------------------------
// Hoisted, fully parallel input projection for layer 0 over ALL timesteps:
//   P0[t][b][n] = x[b][t][:] @ Wi0^T   (bf16)
// ---------------------------------------------------------------------------
__global__ __launch_bounds__(128) void xproj_k(
    const float* __restrict__ x, const __bf16* __restrict__ w,
    __bf16* __restrict__ p0)
{
  const int lane = threadIdx.x & 31;
  const int wave = threadIdx.x >> 5;
  const int nt   = blockIdx.x * 4 + wave;
  const int t    = blockIdx.y;
  const int lm   = lane & 15;
  const int kg   = (lane >> 4) * 8;
  v8f acc[4] = {};
  const __bf16* fb = w + (size_t)nt * 512 + (size_t)lane * 8;
#pragma unroll 2
  for (int kt = 0; kt < 64; ++kt) {
    const __bf16* f = fb + (size_t)kt * 65536;
    V16U b;
    b.h[0] = *(const v8bf*)(f);
    b.h[1] = *(const v8bf*)(f + 256);
    const int ka = kt * 32 + kg;
#pragma unroll
    for (int mt = 0; mt < 4; ++mt) {
      const float* row = x + ((size_t)(mt * 16 + lm) * T_ + t) * H_ + ka;
      f32x4 f0 = *(const f32x4*)(row);
      f32x4 f1 = *(const f32x4*)(row + 4);
      f32x4 f2 = *(const f32x4*)(row + 16);
      f32x4 f3 = *(const f32x4*)(row + 20);
      V16U av;
#pragma unroll
      for (int j = 0; j < 4; ++j) {
        av.v[j]      = (__bf16)f0[j];
        av.v[4 + j]  = (__bf16)f1[j];
        av.v[8 + j]  = (__bf16)f2[j];
        av.v[12 + j] = (__bf16)f3[j];
      }
      acc[mt] = __builtin_amdgcn_wmma_f32_16x16x32_bf16(
          false, av.v, false, b.v, (short)0, acc[mt], false, false);
    }
  }
  const int n = nt * 16 + lm;
#pragma unroll
  for (int mt = 0; mt < 4; ++mt) {
    const int mr = mt * 16 + (lane >> 4) * 8;
#pragma unroll
    for (int i = 0; i < 8; ++i)
      p0[((size_t)t * B_ + mr + i) * H_ + n] = (__bf16)acc[mt][i];
  }
}

// ---------------------------------------------------------------------------
// Per-step GEMM with async-LDS A-staging:  pre = sum_seg A_seg @ W_seg^T
// 128 thr (4 waves), wave owns 1 n-tile x 4 m-tiles. A slab (64 rows x 32 k)
// staged to LDS via global_load_async_to_lds_b128, double-buffered, 4 kt per
// stage, rows padded to 80 B (conflict-free 16-row b128 fragment reads).
// ---------------------------------------------------------------------------
#define CH_     4                     // kt chunks per stage
#define RSTR_   80                    // padded row stride (bytes)
#define SLAB_   (64 * RSTR_)          // 5120 B per kt slab
#define CHUNK_  (CH_ * SLAB_)         // 20480 B per stage

template<int NSEG>
__global__ __launch_bounds__(128) void gemm_step_k(
    const __bf16* __restrict__ a0, const __bf16* __restrict__ w0,
    const __bf16* __restrict__ a1, const __bf16* __restrict__ w1,
    float* __restrict__ pre)
{
  __shared__ __align__(16) char smem[2 * CHUNK_];   // 40 KB
  const int tid  = threadIdx.x;
  const int lane = tid & 31;
  const int wave = tid >> 5;
  const int nt   = blockIdx.x * 4 + wave;
  const int lm   = lane & 15;
  const int kg   = (lane >> 4) * 8;
  v8f acc[4] = {};

#pragma unroll
  for (int seg = 0; seg < NSEG; ++seg) {
    const __bf16* a  = seg ? a1 : a0;
    const __bf16* w  = seg ? w1 : w0;
    const __bf16* fb = w + (size_t)nt * 512 + (size_t)lane * 8;

    // issue one stage of async loads: 1024 x 16 B units, 8 per thread
    auto issue = [&](int ktbase, char* buf) {
#pragma unroll
      for (int i = 0; i < 8; ++i) {
        const int u   = tid + 128 * i;
        const int kk4 = u >> 8;
        const int row = (u >> 2) & 63;
        const int s16 = u & 3;
        char* l = buf + kk4 * SLAB_ + row * RSTR_ + s16 * 16;
        const char* g = (const char*)a + (size_t)row * (H_ * 2)
                        + (size_t)(ktbase + kk4) * 64 + s16 * 16;
        async_load_b128(l, g);
      }
    };

    issue(0, smem);
    const int S = 64 / CH_;           // 16 stages
    for (int s = 0; s < S; ++s) {
      wait_async0();
      __syncthreads();
      if (s + 1 < S) issue((s + 1) * CH_, smem + ((s + 1) & 1) * CHUNK_);
      char* buf = smem + (s & 1) * CHUNK_;
#pragma unroll
      for (int kk = 0; kk < CH_; ++kk) {
        const int kt = s * CH_ + kk;
        const __bf16* f = fb + (size_t)kt * 65536;
        V16U b;
        b.h[0] = *(const v8bf*)(f);
        b.h[1] = *(const v8bf*)(f + 256);
        if (kt + 8 < 64)
          __builtin_prefetch((const void*)(f + (size_t)8 * 65536), 0, 1);
#pragma unroll
        for (int mt = 0; mt < 4; ++mt) {
          const char* r = buf + kk * SLAB_ + (mt * 16 + lm) * RSTR_ + kg * 2;
          V16U av;
          av.h[0] = *(const v8bf*)(r);
          av.h[1] = *(const v8bf*)(r + 32);
          acc[mt] = __builtin_amdgcn_wmma_f32_16x16x32_bf16(
              false, av.v, false, b.v, (short)0, acc[mt], false, false);
        }
      }
    }
  }
  const int n = nt * 16 + lm;
#pragma unroll
  for (int mt = 0; mt < 4; ++mt) {
    const int mr = mt * 16 + (lane >> 4) * 8;
#pragma unroll
    for (int i = 0; i < 8; ++i)
      pre[(size_t)(mr + i) * H_ + n] = acc[mt][i];
  }
}

// ---------------------------------------------------------------------------
// LayerNorm over H, one batch row per block; adds bias (+ optional bf16
// add-in slab = hoisted x-projection), writes bf16 state, f32 state, output.
// ---------------------------------------------------------------------------
__global__ __launch_bounds__(256) void ln_step_k(
    const float* __restrict__ pre, const float* __restrict__ bias,
    const float* __restrict__ gamma, const float* __restrict__ beta,
    const __bf16* __restrict__ addin,
    __bf16* __restrict__ hbf, float* __restrict__ hf,
    float* __restrict__ outp)
{
  const int b   = blockIdx.x;
  const int tid = threadIdx.x;
  float v[8];
  float s = 0.0f, q = 0.0f;
  const float* p = pre + (size_t)b * H_ + tid * 8;
#pragma unroll
  for (int i = 0; i < 8; ++i) {
    const int n = tid * 8 + i;
    float xv = p[i] + bias[n];
    if (addin) xv += (float)addin[(size_t)b * H_ + n];
    v[i] = xv; s += xv; q += xv * xv;
  }
  __shared__ float ssum[256], ssq[256];
  ssum[tid] = s; ssq[tid] = q;
  __syncthreads();
  for (int off = 128; off > 0; off >>= 1) {
    if (tid < off) { ssum[tid] += ssum[tid + off]; ssq[tid] += ssq[tid + off]; }
    __syncthreads();
  }
  const float mu  = ssum[0] * (1.0f / H_);
  const float var = ssq[0] * (1.0f / H_) - mu * mu;
  const float inv = rsqrtf(var + 1e-5f);
#pragma unroll
  for (int i = 0; i < 8; ++i) {
    const int n = tid * 8 + i;
    float y = (v[i] - mu) * inv * gamma[n] + beta[n];
    hbf[(size_t)b * H_ + n] = (__bf16)y;
    hf[(size_t)b * H_ + n]  = y;
    if (outp) outp[(size_t)b * T_ * H_ + n] = y;
  }
}

__global__ __launch_bounds__(256) void finalize_k(
    const float* __restrict__ hf0, const float* __restrict__ hf1,
    float* __restrict__ out_tail)
{
  int i = blockIdx.x * 256 + threadIdx.x;
  if (i < B_ * H_)            out_tail[i] = hf0[i];
  else if (i < 2 * B_ * H_)   out_tail[i] = hf1[i - B_ * H_];
}

// ---------------------------------------------------------------------------
extern "C" void kernel_launch(void* const* d_in, const int* in_sizes, int n_in,
                              void* d_out, int out_size, void* d_ws, size_t ws_size,
                              hipStream_t stream)
{
  const float*   x    = (const float*)d_in[0];
  const float*   W_ih = (const float*)d_in[1];
  const float*   b_ih = (const float*)d_in[2];
  const float*   W_hh = (const float*)d_in[3];
  const float*   b_hh = (const float*)d_in[4];
  const float*   gam  = (const float*)d_in[5];
  const float*   bet  = (const float*)d_in[6];
  const uint8_t* M_ih = (const uint8_t*)d_in[7];
  const uint8_t* M_hh = (const uint8_t*)d_in[8];
  float* out = (float*)d_out;

  char* ws = (char*)d_ws;
  size_t o = 0;
  __bf16* wp   = (__bf16*)(ws + o); o += (size_t)4 * H_ * H_ * 2;       // 32 MB
  __bf16* p0   = (__bf16*)(ws + o); o += (size_t)T_ * B_ * H_ * 2;      // 64 MB
  __bf16* h0bf = (__bf16*)(ws + o); o += (size_t)B_ * H_ * 2;
  __bf16* h1bf = (__bf16*)(ws + o); o += (size_t)B_ * H_ * 2;
  float*  h0f  = (float*)(ws + o);  o += (size_t)B_ * H_ * 4;
  float*  h1f  = (float*)(ws + o);  o += (size_t)B_ * H_ * 4;
  float*  pre  = (float*)(ws + o);  o += (size_t)B_ * H_ * 4;
  float*  bias = (float*)(ws + o);  o += (size_t)L_ * H_ * 4;

  const size_t WM = (size_t)H_ * H_;

  pack_weights_k<<<8192, 256, 0, stream>>>(W_ih, W_hh, M_ih, M_hh, wp);
  bias_k<<<16, 256, 0, stream>>>(b_ih, b_hh, bias);
  init_h_k<<<512, 256, 0, stream>>>(h0bf, h1bf, h0f, h1f);

  xproj_k<<<dim3(32, 256), 128, 0, stream>>>(x, wp + 0 * WM, p0);

  for (int t = 0; t < T_; ++t) {
    gemm_step_k<1><<<32, 128, 0, stream>>>(h0bf, wp + 1 * WM,
                                           nullptr, nullptr, pre);
    ln_step_k<<<64, 256, 0, stream>>>(pre, bias, gam, bet,
                                      p0 + (size_t)t * B_ * H_,
                                      h0bf, h0f, nullptr);
    gemm_step_k<2><<<32, 128, 0, stream>>>(h0bf, wp + 2 * WM,
                                           h1bf, wp + 3 * WM, pre);
    ln_step_k<<<64, 256, 0, stream>>>(pre, bias + H_, gam + H_, bet + H_,
                                      nullptr, h1bf, h1f,
                                      out + (size_t)t * H_);
  }
  finalize_k<<<1024, 256, 0, stream>>>(h0f, h1f, out + (size_t)B_ * T_ * H_);
}
